// Attention_58600533786889
// MI455X (gfx1250) — matrix operand order repaired and checked by reading, not compiled, not run on hardware
//
#include <hip/hip_runtime.h>
#include <math.h>

// ---------------------------------------------------------------------------
// Attention: out = softmax((q Wq^T + bq)(x Wk^T + bk)^T / sqrt(D)) (x Wv^T + bv)
// B=8, Lq=1024, Lk=2048, D=1024.
// All GEMM stages: 128x128 block tiles, async global->LDS double-buffered
// staging (ASYNCcnt), bf16 WMMA (v_wmma_f32_16x16x32_bf16) with f32 accum.
// ---------------------------------------------------------------------------

#define D_DIM 1024
#define LQ    1024
#define LK    2048
#define NB    8

#define BLK        128      // block tile edge (128x128)
#define ROW_BF     40       // staged row: 32 bf16 + 8 pad (80B: 16B-aligned frags,
#define ROW_BYTES  80       //   20-dword stride -> conflict-free ds_load_b128)
#define TILE_BF    (BLK * ROW_BF)

typedef __attribute__((ext_vector_type(4)))  __bf16 bf16x4;
typedef __attribute__((ext_vector_type(8)))  __bf16 bf16x8;
typedef __attribute__((ext_vector_type(16))) __bf16 bf16x16;
typedef __attribute__((ext_vector_type(4)))  float  f32x4;
typedef __attribute__((ext_vector_type(8)))  float  f32x8;

struct __align__(16) SmemTiles {
    __bf16 A[2][TILE_BF];   // 2 x 10 KB
    __bf16 B[2][TILE_BF];   // 2 x 10 KB   (total 40 KB)
};

// ---- WMMA fragment loads from staged LDS rows ------------------------------
// A 16x32 bf16 frag: lane(0-15)=row M, h=lane>>4:
//   V0..3 = K[h*8..h*8+7], V4..7 = K[16+h*8..16+h*8+7]
__device__ __forceinline__ bf16x16 lds_a_frag(const __bf16* row_ptr, int h) {
    bf16x8 lo = *reinterpret_cast<const bf16x8*>(row_ptr + h * 8);
    bf16x8 hi = *reinterpret_cast<const bf16x8*>(row_ptr + 16 + h * 8);
    return __builtin_shufflevector(lo, hi, 0,1,2,3,4,5,6,7,8,9,10,11,12,13,14,15);
}
// B 32x16 bf16 frag: lane(0-15)=col N, h=lane>>4:
//   V0..3 = K[h*8..h*8+7], V4..7 = K[16+h*8..16+h*8+7]
__device__ __forceinline__ bf16x16 lds_b_frag(const __bf16* colrow_ptr) {
    bf16x8 lo = *reinterpret_cast<const bf16x8*>(colrow_ptr);
    bf16x8 hi = *reinterpret_cast<const bf16x8*>(colrow_ptr + 16);
    return __builtin_shufflevector(lo, hi, 0,1,2,3,4,5,6,7,8,9,10,11,12,13,14,15);
}

__device__ __forceinline__ f32x8 wmma_bf16(bf16x16 a, bf16x16 b, f32x8 c) {
    return __builtin_amdgcn_wmma_f32_16x16x32_bf16(
        false, a, false, b, (short)0, c, false, false);
}

// ---------------------------------------------------------------------------
// Block GEMM core: Z[i,j] = sum_k A[i,k]*B[j,k], 128x128 tile per block.
// Ab/Bb are pre-offset to the block's row bases. 8 waves, each 32(i)x64(j).
// 128x32 bf16 panels staged via global_load_async_to_lds_b128, double-buffered.
// ---------------------------------------------------------------------------
__device__ __forceinline__ void gemm_block(
    SmemTiles& sm,
    const __bf16* __restrict__ Ab, int lda,
    const __bf16* __restrict__ Bb, int ldb,
    int klen, f32x8 acc0[4], f32x8 acc1[4]) {
    const int tid  = threadIdx.x;
    const int lane = tid & 31;
    const int lo16 = lane & 15, h = lane >> 4;
    const int wave = tid >> 5;
    const int wi   = (wave & 3) * 32;   // wave's i offset in block
    const int wj   = (wave >> 2) * 64;  // wave's j offset in block

    // Each thread async-copies 4 x 16B chunks per k-step (2 for A, 2 for B).
    auto fill = [&](int buf, int k0) {
        const uint32_t la = (uint32_t)(unsigned long long)&sm.A[buf][0];
        const uint32_t lb = (uint32_t)(unsigned long long)&sm.B[buf][0];
#pragma unroll
        for (int i = 0; i < 2; ++i) {
            const int id = tid + 256 * i;          // 0..511 chunk id
            const int r = id >> 2, q = id & 3;     // row 0..127, 16B quarter
            const __bf16* ga = Ab + (size_t)r * lda + k0 + q * 8;
            asm volatile("global_load_async_to_lds_b128 %0, %1, off"
                         :: "v"(la + r * ROW_BYTES + q * 16), "v"(ga)
                         : "memory");
        }
#pragma unroll
        for (int i = 0; i < 2; ++i) {
            const int id = tid + 256 * i;
            const int r = id >> 2, q = id & 3;
            const __bf16* gb = Bb + (size_t)r * ldb + k0 + q * 8;
            asm volatile("global_load_async_to_lds_b128 %0, %1, off"
                         :: "v"(lb + r * ROW_BYTES + q * 16), "v"(gb)
                         : "memory");
        }
    };

    fill(0, 0);                                    // prologue
    const int nk = klen >> 5;
    for (int ks = 0; ks < nk; ++ks) {
        const int cur = ks & 1;
        if (ks + 1 < nk) {
            fill(cur ^ 1, (ks + 1) << 5);          // prefetch next panel
            // async loads complete in order: oldest 4 (current buffer) done
            asm volatile("s_wait_asynccnt 4" ::: "memory");
        } else {
            asm volatile("s_wait_asynccnt 0" ::: "memory");
        }
        __syncthreads();                           // panel visible to all waves

        const __bf16* sa = &sm.A[cur][0];
        const __bf16* sb = &sm.B[cur][0];
        const bf16x16 a0 = lds_a_frag(sa + (wi + lo16) * ROW_BF, h);
        const bf16x16 a1 = lds_a_frag(sa + (wi + 16 + lo16) * ROW_BF, h);
#pragma unroll
        for (int t = 0; t < 4; ++t) {
            const bf16x16 b = lds_b_frag(sb + (wj + 16 * t + lo16) * ROW_BF + h * 8);
            acc0[t] = wmma_bf16(a0, b, acc0[t]);
            acc1[t] = wmma_bf16(a1, b, acc1[t]);
        }
        __syncthreads();                           // done reading before refill
    }
}

// ---- fp32 -> bf16 conversion ----------------------------------------------
__global__ __launch_bounds__(256) void k_cvt_bf16(const float* __restrict__ in,
                                                  __bf16* __restrict__ out, int n4) {
    int i = blockIdx.x * blockDim.x + threadIdx.x;
    if (i < n4) {
        f32x4 x = *reinterpret_cast<const f32x4*>(in + (size_t)i * 4);
        bf16x4 y;
#pragma unroll
        for (int j = 0; j < 4; ++j) y[j] = (__bf16)x[j];
        *reinterpret_cast<bf16x4*>(out + (size_t)i * 4) = y;
    }
}

// ---- Q/K projection: Z[n,m]=sum_k W[n,k]*X[m,k]; transposed bf16 store -----
// -> Y row-major [seqTotal, D], bias on n (=i).
__global__ __launch_bounds__(256) void k_proj_qk(
    const __bf16* __restrict__ W, const __bf16* __restrict__ X,
    const float* __restrict__ bias, __bf16* __restrict__ Y) {
    __shared__ SmemTiles sm;
    const int i_blk = blockIdx.y * BLK;    // out-feature n
    const int j_blk = blockIdx.x * BLK;    // sequence m
    f32x8 acc0[4] = {}, acc1[4] = {};
    gemm_block(sm, W + (size_t)i_blk * D_DIM, D_DIM,
                   X + (size_t)j_blk * D_DIM, D_DIM, D_DIM, acc0, acc1);
    const int tid = threadIdx.x, wave = tid >> 5, lane = tid & 31;
    const int lo16 = lane & 15, h = lane >> 4;
    const int i0 = i_blk + (wave & 3) * 32;
    const int j0 = j_blk + (wave >> 2) * 64;
    float b0[8], b1[8];
#pragma unroll
    for (int v = 0; v < 8; ++v) {
        b0[v] = bias[i0 + 8 * h + v];
        b1[v] = bias[i0 + 16 + 8 * h + v];
    }
#pragma unroll
    for (int t = 0; t < 4; ++t) {
        bf16x8 p0, p1;
#pragma unroll
        for (int v = 0; v < 8; ++v) {
            p0[v] = (__bf16)(acc0[t][v] + b0[v]);
            p1[v] = (__bf16)(acc1[t][v] + b1[v]);
        }
        __bf16* row = Y + (size_t)(j0 + 16 * t + lo16) * D_DIM;
        *reinterpret_cast<bf16x8*>(row + i0 + 8 * h)      = p0;
        *reinterpret_cast<bf16x8*>(row + i0 + 16 + 8 * h) = p1;
    }
}

// ---- V projection: Z[m,n]=sum_k X[m,k]*Wv[n,k]; transposed store -> Vt -----
// Vt row-major [D, NB*LK], bias on n (=j).
__global__ __launch_bounds__(256) void k_proj_v(
    const __bf16* __restrict__ X, const __bf16* __restrict__ Wv,
    const float* __restrict__ bias, __bf16* __restrict__ Vt) {
    __shared__ SmemTiles sm;
    const int i_blk = blockIdx.y * BLK;    // sequence m (batch-folded)
    const int j_blk = blockIdx.x * BLK;    // out-feature n
    f32x8 acc0[4] = {}, acc1[4] = {};
    gemm_block(sm, X + (size_t)i_blk * D_DIM, D_DIM,
                   Wv + (size_t)j_blk * D_DIM, D_DIM, D_DIM, acc0, acc1);
    const int tid = threadIdx.x, wave = tid >> 5, lane = tid & 31;
    const int lo16 = lane & 15, h = lane >> 4;
    const int i0 = i_blk + (wave & 3) * 32;
    const int j0 = j_blk + (wave >> 2) * 64;
#pragma unroll
    for (int t = 0; t < 4; ++t) {
        const float bj = bias[j0 + 16 * t + lo16];
        bf16x8 p0, p1;
#pragma unroll
        for (int v = 0; v < 8; ++v) {
            p0[v] = (__bf16)(acc0[t][v] + bj);
            p1[v] = (__bf16)(acc1[t][v] + bj);
        }
        __bf16* row = Vt + (size_t)(j0 + 16 * t + lo16) * (NB * LK);
        *reinterpret_cast<bf16x8*>(row + i0 + 8 * h)      = p0;
        *reinterpret_cast<bf16x8*>(row + i0 + 16 + 8 * h) = p1;
    }
}

// ---- scores: S[b][i,j] = scale * sum_d Q[b][i,d]*K[b][j,d] -----------------
__global__ __launch_bounds__(256) void k_scores(
    const __bf16* __restrict__ Q, const __bf16* __restrict__ K,
    float* __restrict__ S, float scale) {
    __shared__ SmemTiles sm;
    const int b = blockIdx.z;
    const __bf16* Qb = Q + (size_t)b * LQ * D_DIM;
    const __bf16* Kb = K + (size_t)b * LK * D_DIM;
    float* Sb = S + (size_t)b * LQ * LK;
    const int i_blk = blockIdx.y * BLK;    // query row
    const int j_blk = blockIdx.x * BLK;    // key index
    f32x8 acc0[4] = {}, acc1[4] = {};
    gemm_block(sm, Qb + (size_t)i_blk * D_DIM, D_DIM,
                   Kb + (size_t)j_blk * D_DIM, D_DIM, D_DIM, acc0, acc1);
    const int tid = threadIdx.x, wave = tid >> 5, lane = tid & 31;
    const int lo16 = lane & 15, h = lane >> 4;
    const int i0 = i_blk + (wave & 3) * 32;
    const int j0 = j_blk + (wave >> 2) * 64;
#pragma unroll
    for (int t = 0; t < 4; ++t)
#pragma unroll
        for (int v = 0; v < 8; ++v) {
            const int jc = j0 + 16 * t + lo16;
            Sb[(size_t)(i0 + 8 * h + v) * LK + jc]      = acc0[t][v] * scale;
            Sb[(size_t)(i0 + 16 + 8 * h + v) * LK + jc] = acc1[t][v] * scale;
        }
}

// ---- softmax over Lk, one wave per row; f32 probs + bf16 copy --------------
__global__ __launch_bounds__(256) void k_softmax(float* __restrict__ S,
                                                 __bf16* __restrict__ P) {
    const int row  = blockIdx.x * (blockDim.x >> 5) + (threadIdx.x >> 5);
    const int lane = threadIdx.x & 31;
    float* srow = S + (size_t)row * LK;
    __bf16* prow = P + (size_t)row * LK;
    float vals[LK / 32];
    float m = -__builtin_inff();
#pragma unroll
    for (int idx = 0; idx < LK / 32; ++idx) {
        float x = srow[lane + idx * 32];
        vals[idx] = x;
        m = fmaxf(m, x);
    }
#pragma unroll
    for (int off = 16; off > 0; off >>= 1) m = fmaxf(m, __shfl_xor(m, off, 32));
    float sum = 0.f;
#pragma unroll
    for (int idx = 0; idx < LK / 32; ++idx) {
        vals[idx] = __expf(vals[idx] - m);
        sum += vals[idx];
    }
#pragma unroll
    for (int off = 16; off > 0; off >>= 1) sum += __shfl_xor(sum, off, 32);
    const float inv = 1.0f / sum;
#pragma unroll
    for (int idx = 0; idx < LK / 32; ++idx) {
        float p = vals[idx] * inv;
        srow[lane + idx * 32] = p;
        prow[lane + idx * 32] = (__bf16)p;
    }
}

// ---- weighted: O[b][i,n] = sum_k P[b][i,k] * Vt[n, b*LK+k] -----------------
__global__ __launch_bounds__(256) void k_av(
    const __bf16* __restrict__ P, const __bf16* __restrict__ Vt,
    float* __restrict__ O) {
    __shared__ SmemTiles sm;
    const int b = blockIdx.z;
    const __bf16* Pb = P + (size_t)b * LQ * LK;
    const __bf16* Vb = Vt + (size_t)b * LK;          // ldb = NB*LK
    float* Ob = O + (size_t)b * LQ * D_DIM;
    const int i_blk = blockIdx.y * BLK;    // query row
    const int j_blk = blockIdx.x * BLK;    // feature n
    f32x8 acc0[4] = {}, acc1[4] = {};
    gemm_block(sm, Pb + (size_t)i_blk * LK, LK,
                   Vb + (size_t)j_blk * (NB * LK), NB * LK, LK, acc0, acc1);
    const int tid = threadIdx.x, wave = tid >> 5, lane = tid & 31;
    const int lo16 = lane & 15, h = lane >> 4;
    const int i0 = i_blk + (wave & 3) * 32;
    const int j0 = j_blk + (wave >> 2) * 64;
#pragma unroll
    for (int t = 0; t < 4; ++t)
#pragma unroll
        for (int v = 0; v < 8; ++v) {
            const int jc = j0 + 16 * t + lo16;
            Ob[(size_t)(i0 + 8 * h + v) * D_DIM + jc]      = acc0[t][v];
            Ob[(size_t)(i0 + 16 + 8 * h + v) * D_DIM + jc] = acc1[t][v];
        }
}

// ---------------------------------------------------------------------------
extern "C" void kernel_launch(void* const* d_in, const int* in_sizes, int n_in,
                              void* d_out, int out_size, void* d_ws, size_t ws_size,
                              hipStream_t stream) {
    const float* x  = (const float*)d_in[0];   // [8,2048,1024]
    const float* q  = (const float*)d_in[1];   // [8,1024,1024]
    const float* Wq = (const float*)d_in[2];
    const float* bq = (const float*)d_in[3];
    const float* Wk = (const float*)d_in[4];
    const float* bk = (const float*)d_in[5];
    const float* Wv = (const float*)d_in[6];
    const float* bv = (const float*)d_in[7];

    float* out_w = (float*)d_out;                       // [8,1024,1024]
    float* out_a = out_w + (size_t)NB * LQ * D_DIM;     // [8,1024,2048]

    char* ws = (char*)d_ws;
    size_t off = 0;
    __bf16* x_bf  = (__bf16*)(ws + off); off += (size_t)NB * LK * D_DIM * 2;
    __bf16* q_bf  = (__bf16*)(ws + off); off += (size_t)NB * LQ * D_DIM * 2;
    __bf16* Wq_bf = (__bf16*)(ws + off); off += (size_t)D_DIM * D_DIM * 2;
    __bf16* Wk_bf = (__bf16*)(ws + off); off += (size_t)D_DIM * D_DIM * 2;
    __bf16* Wv_bf = (__bf16*)(ws + off); off += (size_t)D_DIM * D_DIM * 2;
    __bf16* Q_bf  = (__bf16*)(ws + off); off += (size_t)NB * LQ * D_DIM * 2;
    __bf16* K_bf  = (__bf16*)(ws + off); off += (size_t)NB * LK * D_DIM * 2;
    __bf16* Vt_bf = (__bf16*)(ws + off); off += (size_t)D_DIM * NB * LK * 2;
    __bf16* P_bf  = (__bf16*)(ws + off); off += (size_t)NB * LQ * LK * 2;

    auto cvt = [&](const float* src, __bf16* dst, size_t n) {
        int n4 = (int)(n / 4);
        k_cvt_bf16<<<(n4 + 255) / 256, 256, 0, stream>>>(src, dst, n4);
    };
    cvt(x,  x_bf,  (size_t)NB * LK * D_DIM);
    cvt(q,  q_bf,  (size_t)NB * LQ * D_DIM);
    cvt(Wq, Wq_bf, (size_t)D_DIM * D_DIM);
    cvt(Wk, Wk_bf, (size_t)D_DIM * D_DIM);
    cvt(Wv, Wv_bf, (size_t)D_DIM * D_DIM);

    k_proj_qk<<<dim3(NB * LQ / BLK, D_DIM / BLK), 256, 0, stream>>>(Wq_bf, q_bf, bq, Q_bf);
    k_proj_qk<<<dim3(NB * LK / BLK, D_DIM / BLK), 256, 0, stream>>>(Wk_bf, x_bf, bk, K_bf);
    k_proj_v <<<dim3(D_DIM / BLK, NB * LK / BLK), 256, 0, stream>>>(x_bf, Wv_bf, bv, Vt_bf);

    const float scale = 1.0f / sqrtf((float)D_DIM);
    k_scores<<<dim3(LK / BLK, LQ / BLK, NB), 256, 0, stream>>>(Q_bf, K_bf, out_a, scale);

    k_softmax<<<(NB * LQ) / 8, 256, 0, stream>>>(out_a, P_bf);

    k_av<<<dim3(D_DIM / BLK, LQ / BLK, NB), 256, 0, stream>>>(P_bf, Vt_bf, out_w);
}
